// AttentionBlock_33200097198525
// MI455X (gfx1250) — matrix operand order repaired and checked by reading, not compiled, not run on hardware
//
#include <hip/hip_runtime.h>

typedef __bf16 bf16;
typedef __attribute__((ext_vector_type(16))) __bf16 v16bf;
typedef __attribute__((ext_vector_type(8)))  __bf16 v8bf;
typedef __attribute__((ext_vector_type(8)))  float  v8f;

static constexpr int CCH   = 256;   // channels
static constexpr int NPOS  = 1024;  // H*W
static constexpr int GRP   = 32;    // groups
static constexpr float QSCALE = 0.0625f; // C^-0.5 = 1/16

static __device__ __forceinline__ v16bf join16(v8bf lo, v8bf hi) {
  return __builtin_shufflevector(lo, hi, 0,1,2,3,4,5,6,7,8,9,10,11,12,13,14,15);
}
static __device__ __forceinline__ v8bf  ld8 (const bf16* p) { return *(const v8bf *)p; }
static __device__ __forceinline__ v16bf ld16(const bf16* p) { return *(const v16bf*)p; }

// ---------------- weight f32 -> bf16 ----------------
__global__ void cvt_kernel(const float* a, const float* b, const float* c, const float* d,
                           bf16* oa, bf16* ob, bf16* oc, bf16* od, int n) {
  int i = blockIdx.x * blockDim.x + threadIdx.x;
  if (i < n) { oa[i] = (bf16)a[i]; ob[i] = (bf16)b[i]; oc[i] = (bf16)c[i]; od[i] = (bf16)d[i]; }
}

// ---------------- GroupNorm: x(B,C,N) f32 -> h(B,N,C) bf16 ----------------
__global__ __launch_bounds__(256)
void gn_kernel(const float* __restrict__ x, const float* __restrict__ gw,
               const float* __restrict__ gb, bf16* __restrict__ h) {
  __shared__ float ssum[256];
  __shared__ float ssq[256];
  const int g = blockIdx.x, b = blockIdx.y, tid = threadIdx.x;
  const float* xb = x + (size_t)b * CCH * NPOS + (size_t)g * 8 * NPOS;
  float vals[32];
  float s = 0.f, q = 0.f;
  for (int i = 0; i < 4; ++i) {
    int n = tid + i * 256;
    #pragma unroll
    for (int cc = 0; cc < 8; ++cc) {
      float vv = xb[(size_t)cc * NPOS + n];
      vals[i * 8 + cc] = vv;
      s += vv; q += vv * vv;
    }
  }
  ssum[tid] = s; ssq[tid] = q;
  __syncthreads();
  for (int st = 128; st > 0; st >>= 1) {
    if (tid < st) { ssum[tid] += ssum[tid + st]; ssq[tid] += ssq[tid + st]; }
    __syncthreads();
  }
  float mean = ssum[0] * (1.f / 8192.f);
  float var  = ssq[0]  * (1.f / 8192.f) - mean * mean;
  float rstd = rsqrtf(var + 1e-5f);
  bf16* hb = h + (size_t)b * NPOS * CCH + g * 8;
  for (int i = 0; i < 4; ++i) {
    int n = tid + i * 256;
    #pragma unroll
    for (int cc = 0; cc < 8; ++cc) {
      int c = g * 8 + cc;
      float vv = (vals[i * 8 + cc] - mean) * rstd * gw[c] + gb[c];
      hb[(size_t)n * CCH + cc] = (bf16)vv;
    }
  }
}

// ---------------- projection GEMM: out = W*h + bias ----------------
// MODE 0: out stored (B,N,C) [for qT/kT];  MODE 1: out stored (B,C,N) [for v]
template<int MODE>
__global__ __launch_bounds__(256)
void proj_kernel(const bf16* __restrict__ W, const float* __restrict__ bias,
                 const bf16* __restrict__ h, bf16* __restrict__ out, float scale) {
  const int b = blockIdx.z, tid = threadIdx.x;
  const int wave = tid >> 5, lane = tid & 31, l16 = lane & 15, hf = lane >> 4;
  const int m0 = blockIdx.y * 64 + (wave & 3) * 16;
  const int n0 = blockIdx.x * 32 + (wave >> 2) * 16;
  const bf16* hb   = h + (size_t)b * NPOS * CCH;
  const bf16* wrow = W  + (size_t)(m0 + l16) * CCH;
  const bf16* hrow = hb + (size_t)(n0 + l16) * CCH;
  v16bf af[8], bfr[8];
  #pragma unroll
  for (int f = 0; f < 8; ++f) {
    af[f]  = ld16(wrow + f * 32 + hf * 16);
    bfr[f] = ld16(hrow + f * 32 + hf * 16);
  }
  v8f acc = {};
  #pragma unroll
  for (int f = 0; f < 8; ++f)
    acc = __builtin_amdgcn_wmma_f32_16x16x32_bf16(false, af[f], false, bfr[f], (short)0, acc, false, false);
  bf16* ob = out + (MODE == 1 ? (size_t)b * CCH * NPOS : (size_t)b * NPOS * CCH);
  #pragma unroll
  for (int r = 0; r < 8; ++r) {
    int o = m0 + r + 8 * hf;
    int n = n0 + l16;
    float val = (acc[r] + bias[o]) * scale;
    if (MODE == 1) ob[(size_t)o * NPOS + n] = (bf16)val;
    else           ob[(size_t)n * CCH + o]  = (bf16)val;
  }
}

// ---------------- flash attention: qT(B,N,C) kT(B,N,C) v(B,C,N) -> attnT(B,N,C) ----------------
__global__ __launch_bounds__(256)
void attn_kernel(const bf16* __restrict__ qT, const bf16* __restrict__ kT,
                 const bf16* __restrict__ vC, bf16* __restrict__ attnT) {
  __shared__ bf16  Plds[16][128];
  __shared__ float tileMax[8][16];
  __shared__ float tileSum[8][16];
  __shared__ float rowM[16], rowS[16], rowA[16];

  const int b = blockIdx.y, n0 = blockIdx.x * 16, tid = threadIdx.x;
  const int wave = tid >> 5, lane = tid & 31, l16 = lane & 15, hf = lane >> 4;

  const bf16* q = qT + (size_t)b * NPOS * CCH;
  const bf16* k = kT + (size_t)b * NPOS * CCH;
  const bf16* v = vC + (size_t)b * CCH * NPOS;

  // resident Q A-fragments for rows n0..n0+15, K = 256 (8 frags)
  v16bf qa[8];
  {
    const bf16* qrow = q + (size_t)(n0 + l16) * CCH;
    #pragma unroll
    for (int f = 0; f < 8; ++f)
      qa[f] = ld16(qrow + f * 32 + hf * 16);
  }

  const int c0 = wave * 32;       // this wave's output channel slice
  const bf16* vp0 = v + (size_t)(c0 + l16)      * NPOS;
  const bf16* vp1 = v + (size_t)(c0 + 16 + l16) * NPOS;
  v8f o0 = {}, o1 = {};

  if (tid < 16) { rowM[tid] = -__builtin_inff(); rowS[tid] = 0.f; }

  // software pipeline: K B-fragments for chunk 0 loaded before the loop
  v16bf kb[8];
  {
    const bf16* krow = k + (size_t)(wave * 16 + l16) * CCH;
    #pragma unroll
    for (int f = 0; f < 8; ++f) kb[f] = ld16(krow + f * 32 + hf * 16);
  }
  __syncthreads();

  for (int chunk = 0; chunk < NPOS / 128; ++chunk) {
    // S tile (16 rows x 16 cols), K = 256, operands already in registers
    v8f s = {};
    #pragma unroll
    for (int f = 0; f < 8; ++f)
      s = __builtin_amdgcn_wmma_f32_16x16x32_bf16(false, qa[f], false, kb[f], (short)0, s, false, false);
    // issue next chunk's K loads now -> they drain under softmax + PV phases
    if (chunk + 1 < NPOS / 128) {
      const bf16* krow = k + (size_t)((chunk + 1) * 128 + wave * 16 + l16) * CCH;
      #pragma unroll
      for (int f = 0; f < 8; ++f) kb[f] = ld16(krow + f * 32 + hf * 16);
      // prefetch next chunk's V stripes into caches
      __builtin_prefetch(vp0 + (chunk + 1) * 128 + hf * 16, 0, 1);
      __builtin_prefetch(vp1 + (chunk + 1) * 128 + hf * 16, 0, 1);
    }
    // per-row tile max (cols live across 16-lane halves)
    #pragma unroll
    for (int r = 0; r < 8; ++r) {
      float m = s[r];
      m = fmaxf(m, __shfl_xor(m, 1, 32));
      m = fmaxf(m, __shfl_xor(m, 2, 32));
      m = fmaxf(m, __shfl_xor(m, 4, 32));
      m = fmaxf(m, __shfl_xor(m, 8, 32));
      if (l16 == 0) tileMax[wave][r + 8 * hf] = m;
    }
    __syncthreads();
    if (tid < 16) {
      float nm = rowM[tid];
      #pragma unroll
      for (int w = 0; w < 8; ++w) nm = fmaxf(nm, tileMax[w][tid]);
      rowA[tid] = __expf(rowM[tid] - nm);
      rowM[tid] = nm;
    }
    __syncthreads();
    // P = exp(S - rowmax): store to LDS + row sums
    #pragma unroll
    for (int r = 0; r < 8; ++r) {
      int row = r + 8 * hf;
      float p = __expf(s[r] - rowM[row]);
      Plds[row][wave * 16 + l16] = (bf16)p;
      float su = p;
      su += __shfl_xor(su, 1, 32);
      su += __shfl_xor(su, 2, 32);
      su += __shfl_xor(su, 4, 32);
      su += __shfl_xor(su, 8, 32);
      if (l16 == 0) tileSum[wave][row] = su;
    }
    __syncthreads();
    if (tid < 16) {
      float add = 0.f;
      #pragma unroll
      for (int w = 0; w < 8; ++w) add += tileSum[w][tid];
      rowS[tid] = rowS[tid] * rowA[tid] + add;
    }
    // rescale running O by alpha
    #pragma unroll
    for (int r = 0; r < 8; ++r) {
      float a = rowA[r + 8 * hf];
      o0[r] *= a; o1[r] *= a;
    }
    // PV WMMAs (A = P from LDS, B = V pair per k-step)
    #pragma unroll
    for (int kk = 0; kk < 4; ++kk) {
      const int mg = chunk * 128 + kk * 32 + hf * 16;
      v16bf b0 = ld16(vp0 + mg);
      v16bf b1 = ld16(vp1 + mg);
      v16bf pa = *(const v16bf*)&Plds[l16][kk * 32 + hf * 16];
      o0 = __builtin_amdgcn_wmma_f32_16x16x32_bf16(false, pa, false, b0, (short)0, o0, false, false);
      o1 = __builtin_amdgcn_wmma_f32_16x16x32_bf16(false, pa, false, b1, (short)0, o1, false, false);
    }
    __syncthreads();
  }

  // normalize and store attnT (B,N,C)
  bf16* outp = attnT + (size_t)b * NPOS * CCH;
  #pragma unroll
  for (int r = 0; r < 8; ++r) {
    int row = r + 8 * hf;
    float inv = 1.f / rowS[row];
    size_t base = (size_t)(n0 + row) * CCH;
    outp[base + c0 + l16]      = (bf16)(o0[r] * inv);
    outp[base + c0 + 16 + l16] = (bf16)(o1[r] * inv);
  }
}

// ---------------- output GEMM + bias + residual, f32 out ----------------
__global__ __launch_bounds__(256)
void out_kernel(const bf16* __restrict__ W, const float* __restrict__ bias,
                const bf16* __restrict__ attnT, const float* __restrict__ x,
                float* __restrict__ out) {
  const int b = blockIdx.z, tid = threadIdx.x;
  const int wave = tid >> 5, lane = tid & 31, l16 = lane & 15, hf = lane >> 4;
  const int m0 = blockIdx.y * 64 + (wave & 3) * 16;
  const int n0 = blockIdx.x * 32 + (wave >> 2) * 16;
  const bf16* ab   = attnT + (size_t)b * NPOS * CCH;
  const bf16* wrow = W  + (size_t)(m0 + l16) * CCH;
  const bf16* arow = ab + (size_t)(n0 + l16) * CCH;
  v16bf af[8], bfr[8];
  #pragma unroll
  for (int f = 0; f < 8; ++f) {
    af[f]  = ld16(wrow + f * 32 + hf * 16);
    bfr[f] = ld16(arow + f * 32 + hf * 16);
  }
  v8f acc = {};
  #pragma unroll
  for (int f = 0; f < 8; ++f)
    acc = __builtin_amdgcn_wmma_f32_16x16x32_bf16(false, af[f], false, bfr[f], (short)0, acc, false, false);
  #pragma unroll
  for (int r = 0; r < 8; ++r) {
    int o = m0 + r + 8 * hf;
    int n = n0 + l16;
    size_t idx = (size_t)b * CCH * NPOS + (size_t)o * NPOS + n;
    out[idx] = x[idx] + acc[r] + bias[o];
  }
}

extern "C" void kernel_launch(void* const* d_in, const int* in_sizes, int n_in,
                              void* d_out, int out_size, void* d_ws, size_t ws_size,
                              hipStream_t stream) {
  const float* x   = (const float*)d_in[0];
  const float* gnw = (const float*)d_in[1];
  const float* gnb = (const float*)d_in[2];
  const float* wq  = (const float*)d_in[3]; const float* bq = (const float*)d_in[4];
  const float* wk  = (const float*)d_in[5]; const float* bk = (const float*)d_in[6];
  const float* wv  = (const float*)d_in[7]; const float* bv = (const float*)d_in[8];
  const float* wo  = (const float*)d_in[9]; const float* bo = (const float*)d_in[10];

  const int B = in_sizes[0] / (CCH * NPOS);  // 64

  char* ws = (char*)d_ws;
  size_t off = 0;
  auto alloc = [&](size_t bytes) -> void* {
    void* p = ws + off; off += (bytes + 255) & ~(size_t)255; return p;
  };
  bf16* wqb = (bf16*)alloc((size_t)CCH * CCH * 2);
  bf16* wkb = (bf16*)alloc((size_t)CCH * CCH * 2);
  bf16* wvb = (bf16*)alloc((size_t)CCH * CCH * 2);
  bf16* wob = (bf16*)alloc((size_t)CCH * CCH * 2);
  size_t tsz = (size_t)B * NPOS * CCH * 2;
  bf16* h   = (bf16*)alloc(tsz);   // (B,N,C) normalized input; reused as attnT
  bf16* qT  = (bf16*)alloc(tsz);   // (B,N,C), pre-scaled
  bf16* kTb = (bf16*)alloc(tsz);   // (B,N,C)
  bf16* vCb = (bf16*)alloc(tsz);   // (B,C,N)
  bf16* attnT = h;                 // h dead after projections

  cvt_kernel<<<(CCH * CCH + 255) / 256, 256, 0, stream>>>(wq, wk, wv, wo, wqb, wkb, wvb, wob, CCH * CCH);
  gn_kernel<<<dim3(GRP, B), 256, 0, stream>>>(x, gnw, gnb, h);

  dim3 pgrid(NPOS / 32, CCH / 64, B);
  proj_kernel<0><<<pgrid, 256, 0, stream>>>(wqb, bq, h, qT,  QSCALE);
  proj_kernel<0><<<pgrid, 256, 0, stream>>>(wkb, bk, h, kTb, 1.0f);
  proj_kernel<1><<<pgrid, 256, 0, stream>>>(wvb, bv, h, vCb, 1.0f);

  attn_kernel<<<dim3(NPOS / 16, B), 256, 0, stream>>>(qT, kTb, vCb, attnT);

  out_kernel<<<pgrid, 256, 0, stream>>>(wob, bo, attnT, x, (float*)d_out);
}
